// LGCN_Layer_76785425318243
// MI455X (gfx1250) — compile-verified, matrix operand order checked by reading, not run on hardware
//
#include <hip/hip_runtime.h>
#include <hip/hip_bf16.h>
#include <stdint.h>

// ---------------------------------------------------------------------------
// LGCN layer for MI455X (gfx1250, wave32, WMMA).
// Heavy op: pm_pd (8192x16384 f32, 512MB) GEMMs -> HBM-bound; stream f32,
// convert to bf16 (native v_cvt_pk_bf16_f32) into double-buffered LDS,
// async-copy the bf16 B operand global->LDS (ASYNCcnt),
// v_wmma_f32_16x16x32_bf16 with f32 accumulation.
// ---------------------------------------------------------------------------

typedef __attribute__((ext_vector_type(16))) __bf16 v16bf;
typedef __attribute__((ext_vector_type(8)))  float  v8f;

union BF16x16 { v16bf v; unsigned u[8]; };

__device__ __forceinline__ unsigned pack2bf(float a, float b) {
  unsigned short ul = __builtin_bit_cast(unsigned short, (__bf16)a);
  unsigned short uh = __builtin_bit_cast(unsigned short, (__bf16)b);
  return (unsigned)ul | ((unsigned)uh << 16);
}

__device__ __forceinline__ __bf16 f2bf(float a) { return (__bf16)a; }

// k-pair base inside a bf16 16x32 WMMA fragment for VGPR slot v, lane-half h.
__device__ __forceinline__ int kbase(int v, int h) {
  return ((v & 3) * 2) + ((v >> 2) * 16) + h * 8;
}

// ---------------------------------------------------------------------------
__global__ void zero_kernel(float* __restrict__ p, size_t n) {
  size_t i = (size_t)blockIdx.x * blockDim.x + threadIdx.x;
  size_t stride = (size_t)gridDim.x * blockDim.x;
  for (; i < n; i += stride) p[i] = 0.0f;
}

// out[v] += sum over edges e with dst[e]==v of x[src[e]]  (128 feats/row)
__global__ void spmm_kernel(const int* __restrict__ src, const int* __restrict__ dst,
                            const float* __restrict__ x, float* __restrict__ out, int E) {
  int t = blockIdx.x * blockDim.x + threadIdx.x;
  int e = t >> 5;
  int lane = t & 31;
  if (e >= E) return;
  int s = src[e];
  int d = dst[e];
  float4 v = *(reinterpret_cast<const float4*>(x + (size_t)s * 128) + lane);
  float* o = out + (size_t)d * 128 + lane * 4;
  atomicAdd(o + 0, v.x);
  atomicAdd(o + 1, v.y);
  atomicAdd(o + 2, v.z);
  atomicAdd(o + 3, v.w);
}

// feat [rows][128] f32  ->  out [128][rows] bf16 (so WMMA B-frags read k-contig)
__global__ void transpose_bf16_kernel(const float* __restrict__ in,
                                      __bf16* __restrict__ out, int rows) {
  int t = blockIdx.x * blockDim.x + threadIdx.x;
  if (t >= rows * 128) return;
  int r = t >> 7;
  int c = t & 127;
  out[(size_t)c * rows + r] = f2bf(in[t]);
}

// ---------------------------------------------------------------------------
// Big GEMM: out[m, 0..127] = sum_k A(m,k) * B(k,n)
//  TRANS==0: A(m,k) = PM[m*16384 + k]   (t_g  = pm_pd   @ lg_feature)
//  TRANS==1: A(m,k) = PM[k*16384 + m]   (t_lg = pm_pd^T @ g_feature)
// B supplied pre-transposed/converted: BT bf16 [128][K].
// 64 rows x 128 cols per WG, 8 waves, K-step 32, double-buffered LDS,
// B staged with global_load_async_to_lds_b64 (ASYNCcnt).
// ---------------------------------------------------------------------------
template <int TRANS>
__global__ void __launch_bounds__(256)
gemm_pm_kernel(const float* __restrict__ PM, const __bf16* __restrict__ BT,
               float* __restrict__ out, int K) {
  constexpr int PMLD = 16384;
  constexpr int LDA = 34;   // 68B stride: b32-aligned frag reads, conflict-free
  constexpr int LDB = 36;   // 72B stride: 8B-aligned async b64 destinations
  alignas(16) __shared__ __bf16 As[2][64 * LDA];
  alignas(16) __shared__ __bf16 Bs[2][128 * LDB];

  const int tid   = threadIdx.x;
  const int lane  = tid & 31;
  const int wave  = tid >> 5;
  const int m_sub = wave & 3;   // 16-row slice within the 64-row tile
  const int n_sub = wave >> 2;  // 64-col half
  const int m0    = blockIdx.x * 64;
  const int hi16  = lane >> 4;
  const int l16   = lane & 15;

  // staging work assignment
  const int ar  = tid >> 3;        // 0..31
  const int ac4 = (tid & 7) * 4;   // TRANS==0: k within tile
  const int amc = (tid & 7) * 8;   // TRANS==1: m within tile
  const int bn  = tid >> 1;        // 0..127
  const int bh  = (tid & 1) * 16;  // 0 / 16

  v8f acc[4] = {};

  auto stageA = [&](int k0, __bf16* Ad) {
    if (TRANS == 0) {
#pragma unroll
      for (int rr = 0; rr < 2; ++rr) {
        const float* gp = PM + (size_t)(m0 + ar + rr * 32) * PMLD + k0 + ac4;
        __builtin_prefetch(gp + 64, 0, 0);   // global_prefetch_b8: future k-tiles
        float4 v = *reinterpret_cast<const float4*>(gp);
        unsigned* dp = reinterpret_cast<unsigned*>(&Ad[(ar + rr * 32) * LDA + ac4]);
        dp[0] = pack2bf(v.x, v.y);
        dp[1] = pack2bf(v.z, v.w);
      }
    } else {
      const float* gp = PM + (size_t)(k0 + ar) * PMLD + m0 + amc;
      __builtin_prefetch(gp + 32 * PMLD, 0, 0);
      float4 v0 = *reinterpret_cast<const float4*>(gp);
      float4 v1 = *reinterpret_cast<const float4*>(gp + 4);
      Ad[(amc + 0) * LDA + ar] = f2bf(v0.x);
      Ad[(amc + 1) * LDA + ar] = f2bf(v0.y);
      Ad[(amc + 2) * LDA + ar] = f2bf(v0.z);
      Ad[(amc + 3) * LDA + ar] = f2bf(v0.w);
      Ad[(amc + 4) * LDA + ar] = f2bf(v1.x);
      Ad[(amc + 5) * LDA + ar] = f2bf(v1.y);
      Ad[(amc + 6) * LDA + ar] = f2bf(v1.z);
      Ad[(amc + 7) * LDA + ar] = f2bf(v1.w);
    }
  };

  auto stageB = [&](int k0, __bf16* Bd) {
    // 32B per thread: async copy global bf16 -> LDS, no VGPR round-trip.
    const __bf16* gsrc = BT + (size_t)bn * K + k0 + bh;
    unsigned ldsb = (unsigned)(uintptr_t)&Bd[bn * LDB + bh];
    asm volatile(
        "global_load_async_to_lds_b64 %0, %1, off\n\t"
        "global_load_async_to_lds_b64 %0, %1, off offset:8\n\t"
        "global_load_async_to_lds_b64 %0, %1, off offset:16\n\t"
        "global_load_async_to_lds_b64 %0, %1, off offset:24"
        :: "v"(ldsb), "v"(gsrc) : "memory");
  };

  // prologue: fill buffer 0
  stageA(0, As[0]);
  stageB(0, Bs[0]);
  asm volatile("s_wait_asynccnt 0x0" ::: "memory");
  __syncthreads();

  const int nsteps = K >> 5;
  for (int s = 0; s < nsteps; ++s) {
    const int cur = s & 1;
    if (s + 1 < nsteps) {       // stage next tile into the other buffer
      stageA((s + 1) * 32, As[1 - cur]);
      stageB((s + 1) * 32, Bs[1 - cur]);
    }

    const __bf16* Ac = As[cur];
    const __bf16* Bc = Bs[cur];
    BF16x16 af;
#pragma unroll
    for (int v = 0; v < 8; ++v) {
      af.u[v] = *reinterpret_cast<const unsigned*>(
          &Ac[(m_sub * 16 + l16) * LDA + kbase(v, hi16)]);
    }
#pragma unroll
    for (int tt = 0; tt < 4; ++tt) {
      BF16x16 bf_;
      int n = n_sub * 64 + tt * 16 + l16;
#pragma unroll
      for (int v = 0; v < 8; ++v) {
        bf_.u[v] = *reinterpret_cast<const unsigned*>(&Bc[n * LDB + kbase(v, hi16)]);
      }
      acc[tt] = __builtin_amdgcn_wmma_f32_16x16x32_bf16(
          false, af.v, false, bf_.v, (short)0, acc[tt], false, false);
    }

    asm volatile("s_wait_asynccnt 0x0" ::: "memory");
    __syncthreads();
  }

#pragma unroll
  for (int tt = 0; tt < 4; ++tt) {
    int col = n_sub * 64 + tt * 16 + l16;
#pragma unroll
    for (int r = 0; r < 8; ++r) {
      int row = m0 + m_sub * 16 + r + hi16 * 8;
      out[(size_t)row * 128 + col] = acc[tt][r];
    }
  }
}

// ---------------------------------------------------------------------------
// Fused epilogue GEMM: 16 rows x 128 cols per block (4 waves x 2 tiles):
//   pre[r,o] = sum_{s=0..5} X_s[r,:] @ W_s[o,:]  + total_bias[o]
// X = {hop1, hop2, hop4, feat, deg*feat, t_deg}; partial ReLU (rows<64),
// write pre-BN output, atomic column sum/sumsq for BatchNorm.
// ---------------------------------------------------------------------------
__global__ void __launch_bounds__(128)
fuse_kernel(const float* __restrict__ A0, const float* __restrict__ A1,
            const float* __restrict__ A2, const float* __restrict__ feat,
            const float* __restrict__ deg, const float* __restrict__ Atg,
            const float* __restrict__ Wstack, const float* __restrict__ Wprev,
            const float* __restrict__ Wfuse, const float* __restrict__ Wdeg,
            const float* __restrict__ Bstack, const float* __restrict__ bprev,
            const float* __restrict__ bfuse, const float* __restrict__ bdeg,
            float* __restrict__ outp, float* __restrict__ colsum,
            float* __restrict__ colsq) {
  const int lane = threadIdx.x & 31;
  const int wave = threadIdx.x >> 5;
  const int hi16 = lane >> 4;
  const int l16  = lane & 15;
  const int row0 = blockIdx.x * 16;
  const int arow = row0 + l16;

  const float* Asrc[6] = {A0, A1, A2, feat, feat, Atg};
  const float* Wsrc[6] = {Wstack, Wstack + 128 * 128, Wstack + 2 * 128 * 128,
                          Wprev, Wfuse, Wdeg};

  v8f acc[2] = {};

  for (int s = 0; s < 6; ++s) {
    const float* Ap = Asrc[s];
    const float* Wp = Wsrc[s];
    const float scale = (s == 4) ? deg[arow] : 1.0f;
    for (int k0 = 0; k0 < 128; k0 += 32) {
      BF16x16 af;
#pragma unroll
      for (int v = 0; v < 8; ++v) {
        float2 p = *reinterpret_cast<const float2*>(
            Ap + (size_t)arow * 128 + k0 + kbase(v, hi16));
        af.u[v] = pack2bf(p.x * scale, p.y * scale);
      }
#pragma unroll
      for (int tt = 0; tt < 2; ++tt) {
        int n = wave * 32 + tt * 16 + l16;
        BF16x16 bf_;
#pragma unroll
        for (int v = 0; v < 8; ++v) {
          float2 p = *reinterpret_cast<const float2*>(
              Wp + (size_t)n * 128 + k0 + kbase(v, hi16));
          bf_.u[v] = pack2bf(p.x, p.y);
        }
        acc[tt] = __builtin_amdgcn_wmma_f32_16x16x32_bf16(
            false, af.v, false, bf_.v, (short)0, acc[tt], false, false);
      }
    }
  }

#pragma unroll
  for (int tt = 0; tt < 2; ++tt) {
    int col = wave * 32 + tt * 16 + l16;
    float bias = Bstack[col] + Bstack[128 + col] + Bstack[256 + col] +
                 bprev[col] + bfuse[col] + bdeg[col];
    float s1 = 0.0f, s2 = 0.0f;
#pragma unroll
    for (int r = 0; r < 8; ++r) {
      int row = row0 + r + hi16 * 8;
      float v = acc[tt][r] + bias;
      if (row < 64) v = fmaxf(v, 0.0f);   // _partial_relu: first feature_size/2 ROWS
      outp[(size_t)row * 128 + col] = v;
      s1 += v;
      s2 += v * v;
    }
    atomicAdd(&colsum[col], s1);
    atomicAdd(&colsq[col], s2);
  }
}

// BatchNorm finalize (biased batch stats), in place on d_out region.
__global__ void bn_kernel(float* __restrict__ x, int rows,
                          const float* __restrict__ colsum,
                          const float* __restrict__ colsq,
                          const float* __restrict__ gamma,
                          const float* __restrict__ beta) {
  int t = blockIdx.x * blockDim.x + threadIdx.x;
  if (t >= rows * 128) return;
  int col = t & 127;
  float inv_n = 1.0f / (float)rows;
  float mean = colsum[col] * inv_n;
  float var = colsq[col] * inv_n - mean * mean;
  float v = x[t];
  x[t] = (v - mean) * rsqrtf(var + 1e-5f) * gamma[col] + beta[col];
}

// ---------------------------------------------------------------------------
extern "C" void kernel_launch(void* const* d_in, const int* in_sizes, int n_in,
                              void* d_out, int out_size, void* d_ws, size_t ws_size,
                              hipStream_t stream) {
  constexpr int N = 8192, M = 16384, E_G = 131072, E_LG = 262144;
  const size_t NF = (size_t)N * 128;
  const size_t MF = (size_t)M * 128;

  const int*   g_src   = (const int*)d_in[0];
  const int*   g_dst   = (const int*)d_in[1];
  const int*   lg_src  = (const int*)d_in[2];
  const int*   lg_dst  = (const int*)d_in[3];
  const float* g_feat  = (const float*)d_in[4];
  const float* lg_feat = (const float*)d_in[5];
  const float* g_deg   = (const float*)d_in[6];
  const float* lg_deg  = (const float*)d_in[7];
  const float* pm      = (const float*)d_in[8];
  const float* gW      = (const float*)d_in[9];
  const float* gB      = (const float*)d_in[10];
  const float* lgW     = (const float*)d_in[11];
  const float* lgB     = (const float*)d_in[12];
  const float* prevg_W = (const float*)d_in[13];
  const float* prevg_b = (const float*)d_in[14];
  const float* prevlg_W= (const float*)d_in[15];
  const float* prevlg_b= (const float*)d_in[16];
  const float* degg_W  = (const float*)d_in[17];
  const float* degg_b  = (const float*)d_in[18];
  const float* deglg_W = (const float*)d_in[19];
  const float* deglg_b = (const float*)d_in[20];
  const float* gfuse_W = (const float*)d_in[21];
  const float* gfuse_b = (const float*)d_in[22];
  const float* lgfuse_W= (const float*)d_in[23];
  const float* lgfuse_b= (const float*)d_in[24];
  const float* ba_g    = (const float*)d_in[25];
  const float* ba_b    = (const float*)d_in[26];
  const float* lin_g   = (const float*)d_in[27];
  const float* lin_b   = (const float*)d_in[28];
  (void)in_sizes; (void)n_in; (void)out_size; (void)ws_size;

  // workspace layout (floats)
  float* ws    = (float*)d_ws;
  float* g_h1  = ws;
  float* g_h2  = g_h1 + NF;
  float* g_tmp = g_h2 + NF;
  float* g_h4  = g_tmp + NF;
  float* lg_h1 = g_h4 + NF;
  float* lg_h2 = lg_h1 + MF;
  float* lg_tmp= lg_h2 + MF;
  float* lg_h4 = lg_tmp + MF;
  float* stats = lg_h4 + MF;          // [sum_g 128][sq_g 128][sum_lg 128][sq_lg 128]
  float* t_g   = stats + 512;
  float* t_lg  = t_g + NF;
  __bf16* gfT  = (__bf16*)(t_lg + MF);            // [128][N] bf16
  __bf16* lgfT = gfT + (size_t)128 * N;           // [128][M] bf16

  float* out_g  = (float*)d_out;
  float* out_lg = out_g + NF;

  // 1. zero atomic-accumulated regions (hop buffers + BN stats)
  size_t zeroN = 4 * NF + 4 * MF + 512;
  zero_kernel<<<2048, 256, 0, stream>>>(ws, zeroN);

  // 2. bf16 transposed feature copies (WMMA B operands)
  transpose_bf16_kernel<<<(int)((NF + 255) / 256), 256, 0, stream>>>(g_feat, gfT, N);
  transpose_bf16_kernel<<<(int)((MF + 255) / 256), 256, 0, stream>>>(lg_feat, lgfT, M);

  // 3. SpMM hop chains (hops 1, 2, 4)
  int gb = (E_G * 32) / 256, lb = (E_LG * 32) / 256;
  spmm_kernel<<<gb, 256, 0, stream>>>(g_src, g_dst, g_feat, g_h1, E_G);
  spmm_kernel<<<gb, 256, 0, stream>>>(g_src, g_dst, g_h1, g_h2, E_G);
  spmm_kernel<<<gb, 256, 0, stream>>>(g_src, g_dst, g_h2, g_tmp, E_G);
  spmm_kernel<<<gb, 256, 0, stream>>>(g_src, g_dst, g_tmp, g_h4, E_G);
  spmm_kernel<<<lb, 256, 0, stream>>>(lg_src, lg_dst, lg_feat, lg_h1, E_LG);
  spmm_kernel<<<lb, 256, 0, stream>>>(lg_src, lg_dst, lg_h1, lg_h2, E_LG);
  spmm_kernel<<<lb, 256, 0, stream>>>(lg_src, lg_dst, lg_h2, lg_tmp, E_LG);
  spmm_kernel<<<lb, 256, 0, stream>>>(lg_src, lg_dst, lg_tmp, lg_h4, E_LG);

  // 4. HBM-bound big GEMMs over pm_pd (WMMA bf16, async B staging)
  gemm_pm_kernel<0><<<N / 64, 256, 0, stream>>>(pm, lgfT, t_g, M);   // pm @ lg_feat
  gemm_pm_kernel<1><<<M / 64, 256, 0, stream>>>(pm, gfT, t_lg, N);   // pm^T @ g_feat

  // 5. fused 6-way linear combine + bias + partial ReLU + BN stats
  fuse_kernel<<<N / 16, 128, 0, stream>>>(g_h1, g_h2, g_h4, g_feat, g_deg, t_g,
                                          gW, prevg_W, gfuse_W, degg_W,
                                          gB, prevg_b, gfuse_b, degg_b,
                                          out_g, stats, stats + 128);
  fuse_kernel<<<M / 16, 128, 0, stream>>>(lg_h1, lg_h2, lg_h4, lg_feat, lg_deg, t_lg,
                                          lgW, prevlg_W, lgfuse_W, deglg_W,
                                          lgB, prevlg_b, lgfuse_b, deglg_b,
                                          out_lg, stats + 256, stats + 384);

  // 6. BatchNorm finalize
  bn_kernel<<<(int)((NF + 255) / 256), 256, 0, stream>>>(out_g, N, stats, stats + 128, ba_g, ba_b);
  bn_kernel<<<(int)((MF + 255) / 256), 256, 0, stream>>>(out_lg, M, stats + 256, stats + 384, lin_g, lin_b);
}
